// INN_44976897523903
// MI455X (gfx1250) — matrix operand order repaired and checked by reading, not compiled, not run on hardware
//
#include <hip/hip_runtime.h>
#include <hip/hip_bf16.h>

typedef __attribute__((ext_vector_type(16))) _Float16 v16h;
typedef __attribute__((ext_vector_type(8)))  float    v8f;

#define Dm   256
#define DIm  512
#define Lm   128
#define Nrep 16
#define ROWS (Nrep*Lm)   // 2048

// ---------------------------------------------------------------------------
// WMMA GEMM: C[M, N] = epilogue( A[M,K](lda) @ W[N,K]^T ), one wave computes a
// 16 x (16*NT) strip. K consumed 32/step via v_wmma_f32_16x16x32_f16 with NT
// accumulators sharing one A fragment. Requires M%16==0, N%(16*NT)==0, and
// (when !KG) K%32==0 with 16B-aligned rows. act: 0=none, 1=softplus.
// acc: 1 -> C += result.
// ---------------------------------------------------------------------------
template<int NT, bool KG>
__global__ __launch_bounds__(32)
void gemm_wmma_k(const float* __restrict__ A, const float* __restrict__ W,
                 const float* __restrict__ bias, float* __restrict__ C,
                 int N, int K, int lda, int act, int acc)
{
    const int tilesN = N / (16 * NT);
    const int tm   = blockIdx.x / tilesN;
    const int tn   = blockIdx.x % tilesN;
    const int lane = threadIdx.x;
    const int row0 = tm * 16, col0 = tn * 16 * NT;

    // A 16x32 f16 frag: lanes 0-15 row=lane, elems 0..7 = K[kc..kc+7],
    // elems 8..15 = K[kc+16..kc+23]; lanes 16-31 row=lane-16, K skewed +8.
    const int a_koff = (lane >> 4) * 8;
    // B 32x16 f16 frag: same K order as A (lanes 16-31 skewed +8).
    const int b_koff = (lane >> 4) * 8;

    const float* __restrict__ arow = A + (size_t)(row0 + (lane & 15)) * lda;
    const float* wrow[NT];
    #pragma unroll
    for (int j = 0; j < NT; ++j)
        wrow[j] = W + (size_t)(col0 + j * 16 + (lane & 15)) * K;

    v8f c[NT];
    #pragma unroll
    for (int j = 0; j < NT; ++j) c[j] = (v8f){};

    for (int kc = 0; kc < K; kc += 32) {
        v16h av;
        if (!KG) {
            const float4 a0 = *(const float4*)(arow + kc + a_koff);
            const float4 a1 = *(const float4*)(arow + kc + a_koff + 4);
            const float4 a2 = *(const float4*)(arow + kc + a_koff + 16);
            const float4 a3 = *(const float4*)(arow + kc + a_koff + 20);
            av[0] = (_Float16)a0.x; av[1] = (_Float16)a0.y;
            av[2] = (_Float16)a0.z; av[3] = (_Float16)a0.w;
            av[4] = (_Float16)a1.x; av[5] = (_Float16)a1.y;
            av[6] = (_Float16)a1.z; av[7] = (_Float16)a1.w;
            av[8]  = (_Float16)a2.x; av[9]  = (_Float16)a2.y;
            av[10] = (_Float16)a2.z; av[11] = (_Float16)a2.w;
            av[12] = (_Float16)a3.x; av[13] = (_Float16)a3.y;
            av[14] = (_Float16)a3.z; av[15] = (_Float16)a3.w;
            if (kc + 32 < K) __builtin_prefetch(arow + kc + 32, 0, 3);
        } else {
            // Branch-free tail guard: clamp index, select value (v_cndmask).
            #pragma unroll
            for (int j = 0; j < 8; ++j) {
                const int k1 = kc + a_koff + j;
                const int k2 = kc + 16 + a_koff + j;
                const float v1 = arow[k1 < K ? k1 : 0];
                const float v2 = arow[k2 < K ? k2 : 0];
                av[j]     = (k1 < K) ? (_Float16)v1 : (_Float16)0.f;
                av[j + 8] = (k2 < K) ? (_Float16)v2 : (_Float16)0.f;
            }
        }
        #pragma unroll
        for (int j = 0; j < NT; ++j) {
            v16h bv;
            if (!KG) {
                const float4 b0 = *(const float4*)(wrow[j] + kc + b_koff);
                const float4 b1 = *(const float4*)(wrow[j] + kc + b_koff + 4);
                const float4 b2 = *(const float4*)(wrow[j] + kc + b_koff + 16);
                const float4 b3 = *(const float4*)(wrow[j] + kc + b_koff + 20);
                bv[0]  = (_Float16)b0.x; bv[1]  = (_Float16)b0.y;
                bv[2]  = (_Float16)b0.z; bv[3]  = (_Float16)b0.w;
                bv[4]  = (_Float16)b1.x; bv[5]  = (_Float16)b1.y;
                bv[6]  = (_Float16)b1.z; bv[7]  = (_Float16)b1.w;
                bv[8]  = (_Float16)b2.x; bv[9]  = (_Float16)b2.y;
                bv[10] = (_Float16)b2.z; bv[11] = (_Float16)b2.w;
                bv[12] = (_Float16)b3.x; bv[13] = (_Float16)b3.y;
                bv[14] = (_Float16)b3.z; bv[15] = (_Float16)b3.w;
            } else {
                #pragma unroll
                for (int e = 0; e < 16; ++e) {
                    const int k = kc + b_koff + (e < 8 ? e : e + 8);
                    const float v = wrow[j][k < K ? k : 0];
                    bv[e] = (k < K) ? (_Float16)v : (_Float16)0.f;
                }
            }
            c[j] = __builtin_amdgcn_wmma_f32_16x16x32_f16(
                       false, av, false, bv, (short)0, c[j], false, false);
        }
    }

    // C/D f32 layout: lanes 0-15: VGPR r -> M=row0+r,   N=col+lane
    //                 lanes 16-31: VGPR r -> M=row0+8+r, N=col+lane-16
    const int rbase = row0 + ((lane >> 4) * 8);
    #pragma unroll
    for (int j = 0; j < NT; ++j) {
        const int cn = col0 + j * 16 + (lane & 15);
        const float bval = bias ? bias[cn] : 0.f;
        #pragma unroll
        for (int r = 0; r < 8; ++r) {
            float v = c[j][r] + bval;
            if (act == 1) v = (v > 20.f) ? v : log1pf(__expf(v));  // softplus
            const size_t idx = (size_t)(rbase + r) * N + cn;
            if (acc) v += C[idx];
            C[idx] = v;
        }
    }
}

// ---------------------------------------------------------------------------
// h[n,l,:] = embedding[x[l],:]  (broadcast across 16 replicas)
// ---------------------------------------------------------------------------
__global__ void embed_k(const int* __restrict__ x, const float* __restrict__ emb,
                        float* __restrict__ h)
{
    const int nl = blockIdx.x;          // n*128 + l
    const int l  = nl % Lm;
    const int d  = threadIdx.x;
    h[(size_t)nl * Dm + d] = emb[(size_t)x[l] * Dm + d];
}

// ---------------------------------------------------------------------------
// LayerNorm, one row (D=256) per 256-thread block
// ---------------------------------------------------------------------------
__global__ void layernorm_k(const float* __restrict__ x, const float* __restrict__ w,
                            const float* __restrict__ b, float* __restrict__ y)
{
    const int row = blockIdx.x, t = threadIdx.x;
    __shared__ float red[256];
    const float v = x[(size_t)row * Dm + t];
    red[t] = v; __syncthreads();
    for (int s = 128; s > 0; s >>= 1) { if (t < s) red[t] += red[t + s]; __syncthreads(); }
    const float mean = red[0] * (1.f / Dm);
    __syncthreads();
    const float d = v - mean;
    red[t] = d * d; __syncthreads();
    for (int s = 128; s > 0; s >>= 1) { if (t < s) red[t] += red[t + s]; __syncthreads(); }
    const float var = red[0] * (1.f / Dm);
    y[(size_t)row * Dm + t] = d * rsqrtf(var + 1e-5f) * w[t] + b[t];
}

// ---------------------------------------------------------------------------
// Causal depthwise conv(4) + SiLU on xz[:, :512] -> u
// ---------------------------------------------------------------------------
__global__ void conv_silu_k(const float* __restrict__ xz, const float* __restrict__ cw,
                            const float* __restrict__ cb, float* __restrict__ u)
{
    const int bl = blockIdx.x;          // b*128 + l
    const int bb = bl / Lm, l = bl % Lm;
    const int c  = threadIdx.x;
    float acc = cb[c];
    #pragma unroll
    for (int j = 0; j < 4; ++j) {
        const int t = l - 3 + j;
        if (t >= 0) acc += xz[((size_t)bb * Lm + t) * 1024 + c] * cw[c * 4 + j];
    }
    u[(size_t)bl * DIm + c] = acc / (1.f + __expf(-acc));
}

// ---------------------------------------------------------------------------
// Selective scan: one block per batch replica (16), thread = channel (512).
// State h[16] in registers, B_t/C_t staged in LDS per step. Fuses the
// skip connection and the SiLU(z) gate -> yg.
// ---------------------------------------------------------------------------
__global__ void scan_k(const float* __restrict__ dt, const float* __restrict__ xdbl,
                       const float* __restrict__ u, const float* __restrict__ A_log,
                       const float* __restrict__ skip_D, const float* __restrict__ xz,
                       float* __restrict__ yg)
{
    const int bb = blockIdx.x;
    const int d  = threadIdx.x;
    __shared__ float Bs[16], Cs[16];
    float Av[16], hs[16];
    #pragma unroll
    for (int s = 0; s < 16; ++s) { Av[s] = -__expf(A_log[d * 16 + s]); hs[s] = 0.f; }
    const float sd = skip_D[d];
    for (int t = 0; t < Lm; ++t) {
        const size_t row = (size_t)bb * Lm + t;
        if (d < 16)       Bs[d]      = xdbl[row * 48 + 16 + d];
        else if (d < 32)  Cs[d - 16] = xdbl[row * 48 + 32 + (d - 16)];
        __syncthreads();
        const float dtv = dt[row * DIm + d];
        const float uv  = u[row * DIm + d];
        const float du  = dtv * uv;
        float y = 0.f;
        #pragma unroll
        for (int s = 0; s < 16; ++s) {
            hs[s] = __expf(dtv * Av[s]) * hs[s] + du * Bs[s];
            y += hs[s] * Cs[s];
        }
        y += uv * sd;
        const float zv = xz[row * 1024 + DIm + d];
        yg[row * DIm + d] = y * (zv / (1.f + __expf(-zv)));
        __syncthreads();
    }
}

// (n*128+l) layout -> (l*16+n) layout, and the reverse with residual add
__global__ void nl2ln_k(const float* __restrict__ src, float* __restrict__ dst)
{
    const int r = blockIdx.x, n = r / Lm, l = r % Lm, d = threadIdx.x;
    dst[((size_t)l * Nrep + n) * Dm + d] = src[(size_t)r * Dm + d];
}
__global__ void ln2nl_add_k(const float* __restrict__ src, float* __restrict__ h)
{
    const int r = blockIdx.x, n = r / Lm, l = r % Lm, d = threadIdx.x;
    h[(size_t)r * Dm + d] += src[((size_t)l * Nrep + n) * Dm + d];
}

// ---------------------------------------------------------------------------
// MHA over the 16 replica-tokens at each sequence position l.
// One block per l; q/k/v (16x256 each) + scores in LDS.
// ---------------------------------------------------------------------------
__global__ void attn_k(const float* __restrict__ qkv, float* __restrict__ o)
{
    const int l = blockIdx.x, t = threadIdx.x;
    __shared__ float sq[16 * 256], sk[16 * 256], sv[16 * 256], sc[4 * 16 * 16];
    for (int i = t; i < 16 * 768; i += 256) {
        const int n = i / 768, c = i % 768;
        const float v = qkv[((size_t)l * Nrep + n) * 768 + c];
        if (c < 256)      sq[n * 256 + c]         = v;
        else if (c < 512) sk[n * 256 + (c - 256)] = v;
        else              sv[n * 256 + (c - 512)] = v;
    }
    __syncthreads();
    for (int i = t; i < 1024; i += 256) {            // scores: head,q,k
        const int hh = i >> 8, qi = (i >> 4) & 15, ki = i & 15;
        float acc = 0.f;
        for (int d2 = 0; d2 < 64; ++d2)
            acc += sq[qi * 256 + hh * 64 + d2] * sk[ki * 256 + hh * 64 + d2];
        sc[i] = acc * 0.125f;                        // 1/sqrt(64)
    }
    __syncthreads();
    if (t < 64) {                                    // softmax per (head,q)
        const int base = t * 16;
        float mx = sc[base];
        for (int k2 = 1; k2 < 16; ++k2) mx = fmaxf(mx, sc[base + k2]);
        float sum = 0.f;
        for (int k2 = 0; k2 < 16; ++k2) { const float e = __expf(sc[base + k2] - mx); sc[base + k2] = e; sum += e; }
        const float inv = 1.f / sum;
        for (int k2 = 0; k2 < 16; ++k2) sc[base + k2] *= inv;
    }
    __syncthreads();
    for (int i = t; i < 4096; i += 256) {            // o = att @ v
        const int qi = i >> 8, c = i & 255, hh = c >> 6;
        float acc = 0.f;
        for (int k2 = 0; k2 < 16; ++k2)
            acc += sc[(hh * 16 + qi) * 16 + k2] * sv[k2 * 256 + c];
        o[((size_t)l * Nrep + qi) * Dm + c] = acc;
    }
}

__global__ void mean_k(const float* __restrict__ h, float* __restrict__ hm)
{
    const int l = blockIdx.x, d = threadIdx.x;
    float acc = 0.f;
    for (int n = 0; n < Nrep; ++n) acc += h[((size_t)n * Lm + l) * Dm + d];
    hm[(size_t)l * Dm + d] = acc * (1.f / Nrep);
}

// ---------------------------------------------------------------------------
extern "C" void kernel_launch(void* const* d_in, const int* in_sizes, int n_in,
                              void* d_out, int out_size, void* d_ws, size_t ws_size,
                              hipStream_t stream)
{
    const int*   x          = (const int*)  d_in[0];
    const float* embedding  = (const float*)d_in[1];
    const float* norm1_w    = (const float*)d_in[2];
    const float* norm1_b    = (const float*)d_in[3];
    const float* norm2_w    = (const float*)d_in[4];
    const float* norm2_b    = (const float*)d_in[5];
    const float* in_proj_w  = (const float*)d_in[6];
    const float* conv_w     = (const float*)d_in[7];
    const float* conv_b     = (const float*)d_in[8];
    const float* x_proj_w   = (const float*)d_in[9];
    const float* dt_proj_w  = (const float*)d_in[10];
    const float* dt_proj_b  = (const float*)d_in[11];
    const float* A_log      = (const float*)d_in[12];
    const float* skip_D     = (const float*)d_in[13];
    const float* out_proj_w = (const float*)d_in[14];
    const float* attn_in_w  = (const float*)d_in[15];
    const float* attn_in_b  = (const float*)d_in[16];
    const float* attn_out_w = (const float*)d_in[17];
    const float* attn_out_b = (const float*)d_in[18];
    const float* norm_f_w   = (const float*)d_in[19];
    const float* norm_f_b   = (const float*)d_in[20];
    const float* head_b     = (const float*)d_in[21];

    float* ws = (float*)d_ws;
    size_t o = 0;
    float* h    = ws + o; o += (size_t)ROWS * Dm;
    float* xn   = ws + o; o += (size_t)ROWS * Dm;
    float* xz   = ws + o; o += (size_t)ROWS * 1024;
    float* u    = ws + o; o += (size_t)ROWS * DIm;
    float* xdbl = ws + o; o += (size_t)ROWS * 48;
    float* dtb  = ws + o; o += (size_t)ROWS * DIm;
    float* yg   = ws + o; o += (size_t)ROWS * DIm;
    float* xc   = ws + o; o += (size_t)ROWS * Dm;
    float* qkv  = ws + o; o += (size_t)ROWS * 768;
    float* xo   = ws + o; o += (size_t)ROWS * Dm;
    float* ao   = ws + o; o += (size_t)ROWS * Dm;
    float* hm   = ws + o; o += (size_t)Lm * Dm;
    float* hn   = ws + o; o += (size_t)Lm * Dm;

    embed_k<<<ROWS, 256, 0, stream>>>(x, embedding, h);

    for (int i = 0; i < 6; ++i) {
        // ---- Mamba block ----
        layernorm_k<<<ROWS, 256, 0, stream>>>(h, norm1_w + i * Dm, norm1_b + i * Dm, xn);
        gemm_wmma_k<4, false><<<(ROWS/16)*(1024/64), 32, 0, stream>>>(
            xn, in_proj_w + (size_t)i * 1024 * Dm, nullptr, xz, 1024, Dm, Dm, 0, 0);
        conv_silu_k<<<ROWS, DIm, 0, stream>>>(xz, conv_w + i * DIm * 4, conv_b + i * DIm, u);
        gemm_wmma_k<3, false><<<(ROWS/16)*(48/48), 32, 0, stream>>>(
            u, x_proj_w + (size_t)i * 48 * DIm, nullptr, xdbl, 48, DIm, DIm, 0, 0);
        gemm_wmma_k<4, true><<<(ROWS/16)*(DIm/64), 32, 0, stream>>>(
            xdbl, dt_proj_w + (size_t)i * DIm * 16, dt_proj_b + i * DIm, dtb,
            DIm, 16, 48, /*softplus*/1, 0);
        scan_k<<<Nrep, DIm, 0, stream>>>(
            dtb, xdbl, u, A_log + (size_t)i * DIm * 16, skip_D + i * DIm, xz, yg);
        gemm_wmma_k<4, false><<<(ROWS/16)*(Dm/64), 32, 0, stream>>>(   // residual fused
            yg, out_proj_w + (size_t)i * Dm * DIm, nullptr, h, Dm, DIm, DIm, 0, /*acc*/1);

        // ---- Cross-replica attention ----
        layernorm_k<<<ROWS, 256, 0, stream>>>(h, norm2_w + i * Dm, norm2_b + i * Dm, xn);
        nl2ln_k<<<ROWS, 256, 0, stream>>>(xn, xc);
        gemm_wmma_k<4, false><<<(ROWS/16)*(768/64), 32, 0, stream>>>(
            xc, attn_in_w + (size_t)i * 768 * Dm, attn_in_b + i * 768, qkv,
            768, Dm, Dm, 0, 0);
        attn_k<<<Lm, 256, 0, stream>>>(qkv, xo);
        gemm_wmma_k<4, false><<<(ROWS/16)*(Dm/64), 32, 0, stream>>>(
            xo, attn_out_w + (size_t)i * Dm * Dm, attn_out_b + i * Dm, ao,
            Dm, Dm, Dm, 0, 0);
        ln2nl_add_k<<<ROWS, 256, 0, stream>>>(ao, h);
    }

    // ---- Head ----
    mean_k<<<Lm, 256, 0, stream>>>(h, hm);
    layernorm_k<<<Lm, 256, 0, stream>>>(hm, norm_f_w, norm_f_b, hn);
    gemm_wmma_k<4, false><<<(Lm/16)*(32000/64), 32, 0, stream>>>(
        hn, embedding, head_b, (float*)d_out, 32000, Dm, Dm, 0, 0);
}
